// SwinTransformerBlock3D_v2_63247688401322
// MI455X (gfx1250) — compile-verified, matrix operand order checked by reading, not run on hardware
//
#include <hip/hip_runtime.h>
#include <hip/hip_bf16.h>
#include <math.h>

typedef __attribute__((ext_vector_type(16))) _Float16 v16h;
typedef __attribute__((ext_vector_type(8)))  _Float16 v8h;
typedef __attribute__((ext_vector_type(8)))  float    v8f;

// ---------------- problem constants ----------------
constexpr int  C_    = 192;
constexpr int  HEADS = 6;
constexpr int  DHD   = 32;      // head dim
constexpr int  NTOK  = 256;     // tokens per window
constexpr int  NWIN  = 256;     // B * nW  (2 * 128)
constexpr int  MTOT  = 65536;   // B*T*H*W
// ---------------- workspace layout (bytes) ----------------
constexpr size_t O_QKVW = 0;                           // 576x192 f16
constexpr size_t O_PROJW = O_QKVW  + (size_t)576*192*2;
constexpr size_t O_FC1W  = O_PROJW + (size_t)192*192*2;
constexpr size_t O_FC2W  = O_FC1W  + (size_t)768*192*2;
constexpr size_t O_TBL   = O_FC2W  + (size_t)192*768*2; // 1575*6 f32
constexpr size_t O_D     = (size_t)1 << 20;
constexpr size_t SZ_QN   = (size_t)NWIN*HEADS*NTOK*DHD*2;   // 25165824
constexpr size_t O_Q   = O_D;
constexpr size_t O_K   = O_D + SZ_QN;
constexpr size_t O_V   = O_D + 2*SZ_QN;                 // v stored d-major (vT)
constexpr size_t O_B   = O_D + 3*SZ_QN;                 // xw16, later attn_cat16
constexpr size_t O_E   = O_B + SZ_QN;                   // proj_out f32, later h2 f32
constexpr size_t O_F   = O_E + (size_t)MTOT*C_*4;       // x1 f32
constexpr size_t O_F16 = O_F + (size_t)MTOT*C_*4;       // x1 f16
constexpr size_t O_H1  = O_D;                           // h1 f16 overlays q/k/v/xw

// LDS byte offset of a shared-memory pointer: low 32 bits of the LDS-aperture
// flat address are the LDS address (ISA 10.2 aperture mapping).
static __device__ __forceinline__ uint32_t lds_off(const void* p) {
  return (uint32_t)(uintptr_t)p;
}
// Async DMA: copy 16B from global to LDS (per lane). Tracked by ASYNCcnt.
static __device__ __forceinline__ void async_b128(uint32_t lds, const void* gaddr) {
  asm volatile("global_load_async_to_lds_b128 %0, %1, off"
               :: "v"(lds), "v"((uint64_t)(uintptr_t)gaddr) : "memory");
}
// Same operands, copy bytes [16,32) of both sides via the instruction offset.
static __device__ __forceinline__ void async_b128x2(uint32_t lds, const void* gaddr) {
  uint64_t g = (uint64_t)(uintptr_t)gaddr;
  asm volatile("global_load_async_to_lds_b128 %0, %1, off\n\t"
               "global_load_async_to_lds_b128 %0, %1, off offset:16"
               :: "v"(lds), "v"(g) : "memory");
}
static __device__ __forceinline__ void wait_async_le3() {
  asm volatile("s_wait_asynccnt 0x3" ::: "memory");
}
static __device__ __forceinline__ void wait_async_0() {
  asm volatile("s_wait_asynccnt 0x0" ::: "memory");
}

// Build a WMMA 16-bit A/B operand from two contiguous 8-half octets in LDS.
// Operand layout: lane holds row (A) / col (B) = lane&15; elements [0..7] are
// K = base..base+7, elements [8..15] are K = 16+base..16+base+7, base=(lane>>4)*8.
static __device__ __forceinline__ v16h make_op(const _Float16* lo, const _Float16* hi) {
  v8h a = *(const v8h*)lo;
  v8h b = *(const v8h*)hi;
  v16h r;
#pragma unroll
  for (int i = 0; i < 8; i++) { r[i] = a[i]; r[i + 8] = b[i]; }
  return r;
}

// ---------------- small helper kernels ----------------
__global__ void k_f32_to_f16(const float* __restrict__ in, _Float16* __restrict__ out, int n) {
  int i = blockIdx.x * blockDim.x + threadIdx.x;
  if (i < n) out[i] = (_Float16)in[i];
}

// continuous position bias table: tbl[m*6+h] = 16*sigmoid(cpb MLP(rel_coords(m)))
__global__ void k_cpb(const float* __restrict__ w1, const float* __restrict__ b1,
                      const float* __restrict__ w2, float* __restrict__ tbl) {
  int idx = blockIdx.x * blockDim.x + threadIdx.x;
  if (idx >= 1575 * 6) return;
  int m = idx / 6, hd = idx % 6;
  int it = m / 225, ih = (m / 15) % 15, iw = m % 15;
  float c[3] = { (float)(it - 3) * (8.0f / 3.0f),
                 (float)(ih - 7) * (8.0f / 7.0f),
                 (float)(iw - 7) * (8.0f / 7.0f) };
  const float inv_log2_8 = 1.0f / log2f(8.0f);
#pragma unroll
  for (int k = 0; k < 3; k++) {
    float x = c[k];
    float s = (x > 0.f) ? 1.f : ((x < 0.f) ? -1.f : 0.f);
    c[k] = s * log2f(fabsf(x) + 1.0f) * inv_log2_8;
  }
  float acc = 0.f;
  for (int j = 0; j < 512; j++) {
    float hsum = w1[j*3+0]*c[0] + w1[j*3+1]*c[1] + w1[j*3+2]*c[2] + b1[j];
    hsum = fmaxf(hsum, 0.f);
    acc += hsum * w2[hd*512 + j];
  }
  tbl[idx] = 16.0f / (1.0f + __expf(-acc));
}

// cyclic shift (-2,-4,-4) + window partition, f32 -> f16
__global__ void k_shift_partition(const float* __restrict__ x, _Float16* __restrict__ xw) {
  int idx = blockIdx.x * blockDim.x + threadIdx.x;     // NWIN*NTOK*C
  int c = idx % 192;
  int n = (idx / 192) & 255;
  int bw = idx / (192 * 256);
  int b = bw >> 7, widx = bw & 127;
  int tb = widx >> 6, hb = (widx >> 3) & 7, wb = widx & 7;
  int dt = n >> 6, dh = (n >> 3) & 7, dw = n & 7;
  int t = (tb*4 + dt + 2) & 7;
  int hh = (hb*8 + dh + 4) & 63;
  int ww = (wb*8 + dw + 4) & 63;
  xw[idx] = (_Float16)x[((((size_t)b*8 + t)*64 + hh)*64 + ww)*192 + c];
}

// L2-normalize rows of q16 and k16 (row = 32 halfs), eps 1e-12
__global__ void k_norm_qk(_Float16* q16, _Float16* k16) {
  constexpr int ROWS = NWIN * HEADS * NTOK;
  int idx = blockIdx.x * blockDim.x + threadIdx.x;
  if (idx >= 2 * ROWS) return;
  _Float16* r = (idx < ROWS) ? (q16 + (size_t)idx * DHD) : (k16 + (size_t)(idx - ROWS) * DHD);
  float s = 0.f;
#pragma unroll
  for (int i = 0; i < DHD; i++) { float v = (float)r[i]; s += v * v; }
  float inv = 1.0f / fmaxf(sqrtf(s), 1e-12f);
#pragma unroll
  for (int i = 0; i < DHD; i++) r[i] = (_Float16)((float)r[i] * inv);
}

// ---------------- generic WMMA GEMM: out = A(MxK,f16) * W(NxK,f16)^T + bias ----
// block tile 128x64, 256 threads (8 waves); wave w -> rows 16w..16w+15, all 64 cols.
// Double-buffered LDS tiles filled by async global->LDS DMA (ASYNCcnt pipeline).
// MODE 0: f32 out; MODE 2: GELU -> f16 out; MODE 3: qkv scatter -> q/k/vT f16
template <int MODE>
__global__ __launch_bounds__(256)
void k_gemm_wmma(const _Float16* __restrict__ A, const _Float16* __restrict__ W,
                 const float* __restrict__ bias, float* __restrict__ outF,
                 _Float16* __restrict__ outH, _Float16* __restrict__ q16,
                 _Float16* __restrict__ k16, _Float16* __restrict__ vT16,
                 int Nn, int K) {
  __shared__ _Float16 As[2][128 * 32];
  __shared__ _Float16 Bs[2][64 * 32];
  int tid = threadIdx.x, lane = tid & 31, wv = tid >> 5;
  int bm = blockIdx.x * 128, bn = blockIdx.y * 64;
  int r = tid >> 2;              // 0..63
  int cc = (tid & 3) * 8;        // 0,8,16,24 (halfs)

  auto issue_tile = [&](int buf, int k0) {
    async_b128(lds_off(&As[buf][r * 32 + cc]),        A + (size_t)(bm + r)      * K + k0 + cc);
    async_b128(lds_off(&As[buf][(r + 64) * 32 + cc]), A + (size_t)(bm + r + 64) * K + k0 + cc);
    async_b128(lds_off(&Bs[buf][r * 32 + cc]),        W + (size_t)(bn + r)      * K + k0 + cc);
  };

  v8f acc[4] = {};
  int rloc = lane & 15, half = lane >> 4, kb = half << 3;
  int row = (wv << 4) + rloc;

  issue_tile(0, 0);
  int cur = 0;
  for (int k0 = 0; k0 < K; k0 += 32) {
    if (k0 + 32 < K) {
      issue_tile(cur ^ 1, k0 + 32);  // prefetch next tile via DMA
      wait_async_le3();              // in-order completion: current tile landed
    } else {
      wait_async_0();
    }
    __syncthreads();                 // all waves' DMAs visible
    v16h a = make_op(&As[cur][row*32 + kb], &As[cur][row*32 + 16 + kb]);
#pragma unroll
    for (int nt = 0; nt < 4; nt++) {
      int wr = (nt << 4) + rloc;
      v16h b = make_op(&Bs[cur][wr*32 + kb], &Bs[cur][wr*32 + 16 + kb]);
      acc[nt] = __builtin_amdgcn_wmma_f32_16x16x32_f16(false, a, false, b,
                                                       (short)0, acc[nt], false, false);
    }
    __syncthreads();                 // tile fully consumed before its buffer is refilled
    cur ^= 1;
  }
  // epilogue: C layout — VGPR e holds row (e + 8*half), col = lane&15
  int mbase = bm + (wv << 4) + (half << 3);
#pragma unroll
  for (int nt = 0; nt < 4; nt++) {
    int gn = bn + (nt << 4) + rloc;
    float bv = bias[gn];
#pragma unroll
    for (int e = 0; e < 8; e++) {
      int gm = mbase + e;
      float val = acc[nt][e] + bv;
      if (MODE == 0) {
        outF[(size_t)gm * Nn + gn] = val;
      } else if (MODE == 2) {
        float g = 0.5f * val * (1.0f + erff(val * 0.70710678118f));
        outH[(size_t)gm * Nn + gn] = (_Float16)g;
      } else { // qkv scatter
        int which = gn / 192, c = gn % 192;
        int hh = c >> 5, dd = c & 31;
        int bw = gm >> 8, n = gm & 255;
        size_t p = (size_t)(bw * 6 + hh);
        if (which == 0)      q16[(p * 256 + n) * 32 + dd] = (_Float16)val;
        else if (which == 1) k16[(p * 256 + n) * 32 + dd] = (_Float16)val;
        else                 vT16[(p * 32 + dd) * 256 + n] = (_Float16)val;
      }
    }
  }
}

// ---------------- fused attention: one block per (window, head) ----------------
// 512 threads = 16 waves; wave w owns query rows 16w..16w+15.
__global__ __launch_bounds__(512)
void k_attn(const _Float16* __restrict__ q16, const _Float16* __restrict__ k16,
            const _Float16* __restrict__ vT16, const float* __restrict__ logit_scale,
            const float* __restrict__ tbl, _Float16* __restrict__ attn_cat) {
  extern __shared__ _Float16 sm[];
  _Float16* qs = sm;             // 256x32
  _Float16* ks = sm + 8192;      // 256x32
  _Float16* vs = sm + 16384;     // 32x256 (d-major)
  _Float16* Ps = sm + 24576;     // 256x256
  int tid = threadIdx.x, lane = tid & 31, wv = tid >> 5;
  int bw = blockIdx.x / 6, h = blockIdx.x % 6;
  size_t base = (size_t)(bw * 6 + h) * (256 * 32);
  // stage q/k/v via async DMA: 32B per thread per array (2 x b128 via offset)
  {
    int o = tid * 16;  // halfs
    async_b128x2(lds_off(qs + o), q16 + base + o);
    async_b128x2(lds_off(ks + o), k16 + base + o);
    async_b128x2(lds_off(vs + o), vT16 + base + o);
  }
  float scale = __expf(fminf(logit_scale[h], 4.6051701860f)); // clamp log(100)
  wait_async_0();
  __syncthreads();

  int r0 = wv << 4;
  int rloc = lane & 15, half = lane >> 4, kb = half << 3;
  int rowq = r0 + rloc;
  // ---- phase 1: S = (qn kn^T), 16 tiles of 16x16 per wave ----
  v16h a = make_op(&qs[rowq*32 + kb], &qs[rowq*32 + 16 + kb]);
  v8f acc[16];
#pragma unroll
  for (int jt = 0; jt < 16; jt++) {
    int rowk = (jt << 4) + rloc;
    v16h b = make_op(&ks[rowk*32 + kb], &ks[rowk*32 + 16 + kb]);
    v8f z = {};
    acc[jt] = __builtin_amdgcn_wmma_f32_16x16x32_f16(false, a, false, b, (short)0, z, false, false);
  }
  // ---- bias + shift-mask + softmax ----
  int widx = bw & 127;
  int tb = widx >> 6, hb = (widx >> 3) & 7, wb = widx & 7;
  int jcol = rloc; // this thread's column within each tile
#pragma unroll
  for (int e = 0; e < 8; e++) {
    int i = r0 + (half << 3) + e;
    int ti = i >> 6, hi = (i >> 3) & 7, wi = i & 7;
    int gt = tb*4 + ti, gh = hb*8 + hi, gw = wb*8 + wi;
    int cnti = (gt < 4 ? 0 : (gt < 6 ? 1 : 2)) * 9 +
               (gh < 56 ? 0 : (gh < 60 ? 1 : 2)) * 3 +
               (gw < 56 ? 0 : (gw < 60 ? 1 : 2));
    float vals[16];
    float m = -1e30f;
#pragma unroll
    for (int jt = 0; jt < 16; jt++) {
      int j = (jt << 4) + jcol;
      int tj = j >> 6, hj = (j >> 3) & 7, wj = j & 7;
      int gt2 = tb*4 + tj, gh2 = hb*8 + hj, gw2 = wb*8 + wj;
      int cntj = (gt2 < 4 ? 0 : (gt2 < 6 ? 1 : 2)) * 9 +
                 (gh2 < 56 ? 0 : (gh2 < 60 ? 1 : 2)) * 3 +
                 (gw2 < 56 ? 0 : (gw2 < 60 ? 1 : 2));
      int rpi = (ti - tj + 3) * 225 + (hi - hj + 7) * 15 + (wi - wj + 7);
      float v = acc[jt][e] * scale + tbl[rpi * 6 + h] + (cnti == cntj ? 0.f : -100.f);
      vals[jt] = v;
      m = fmaxf(m, v);
    }
    for (int s = 1; s < 16; s <<= 1) m = fmaxf(m, __shfl_xor(m, s, 32));
    float sum = 0.f;
#pragma unroll
    for (int jt = 0; jt < 16; jt++) { vals[jt] = __expf(vals[jt] - m); sum += vals[jt]; }
    for (int s = 1; s < 16; s <<= 1) sum += __shfl_xor(sum, s, 32);
    float inv = 1.0f / sum;
#pragma unroll
    for (int jt = 0; jt < 16; jt++)
      Ps[i * 256 + (jt << 4) + jcol] = (_Float16)(vals[jt] * inv);
  }
  __syncthreads();
  // ---- phase 2: out = P * v  (K=256, 2 output tiles of 16 cols) ----
  v8f oacc[2] = {};
  int rowp = r0 + rloc;
#pragma unroll
  for (int kk = 0; kk < 8; kk++) {
    v16h ap = make_op(&Ps[rowp*256 + (kk << 5) + kb], &Ps[rowp*256 + (kk << 5) + 16 + kb]);
#pragma unroll
    for (int nt = 0; nt < 2; nt++) {
      int dd = (nt << 4) + rloc;
      v16h bv = make_op(&vs[dd*256 + (kk << 5) + kb], &vs[dd*256 + (kk << 5) + 16 + kb]);
      oacc[nt] = __builtin_amdgcn_wmma_f32_16x16x32_f16(false, ap, false, bv,
                                                        (short)0, oacc[nt], false, false);
    }
  }
#pragma unroll
  for (int nt = 0; nt < 2; nt++) {
    int dd = (nt << 4) + rloc;
#pragma unroll
    for (int e = 0; e < 8; e++) {
      int tok = r0 + (half << 3) + e;
      attn_cat[((size_t)(bw * 256 + tok)) * 192 + h * 32 + dd] = (_Float16)oacc[nt][e];
    }
  }
}

// ---------------- window-reverse + roll + res-post-norm LN1 ----------------
__global__ __launch_bounds__(192)
void k_ln1(const float* __restrict__ proj_out, const float* __restrict__ x,
           const float* __restrict__ g, const float* __restrict__ bt,
           float* __restrict__ x1, _Float16* __restrict__ x1h) {
  __shared__ float red[192];
  __shared__ float red2[192];
  int tok = blockIdx.x, c = threadIdx.x;
  int b = tok >> 15, t = (tok >> 12) & 7, hh = (tok >> 6) & 63, ww = tok & 63;
  int ts = (t + 6) & 7, hs = (hh + 60) & 63, ws = (ww + 60) & 63; // inverse roll
  int tb = ts >> 2, dt = ts & 3, hb = hs >> 3, dh = hs & 7, wb = ws >> 3, dw = ws & 7;
  int bw = b * 128 + ((tb << 3) + hb) * 8 + wb;
  int n = ((dt << 3) + dh) * 8 + dw;
  float v = proj_out[((size_t)(bw * 256 + n)) * 192 + c];
  red[c] = v; red2[c] = v * v;
  __syncthreads();
  for (int s = 96; s >= 3; s >>= 1) {
    if (c < s) { red[c] += red[c + s]; red2[c] += red2[c + s]; }
    __syncthreads();
  }
  float mean = (red[0] + red[1] + red[2]) * (1.0f / 192.0f);
  float var  = (red2[0] + red2[1] + red2[2]) * (1.0f / 192.0f) - mean * mean;
  float rstd = rsqrtf(var + 1e-5f);
  float o = x[(size_t)tok * 192 + c] + (v - mean) * rstd * g[c] + bt[c];
  x1[(size_t)tok * 192 + c] = o;
  x1h[(size_t)tok * 192 + c] = (_Float16)o;
}

// ---------------- final LN residual -> f32 output ----------------
__global__ __launch_bounds__(192)
void k_ln2(const float* __restrict__ h2, const float* __restrict__ x1,
           const float* __restrict__ g, const float* __restrict__ bt,
           float* __restrict__ out) {
  __shared__ float red[192];
  __shared__ float red2[192];
  int tok = blockIdx.x, c = threadIdx.x;
  float v = h2[(size_t)tok * 192 + c];
  red[c] = v; red2[c] = v * v;
  __syncthreads();
  for (int s = 96; s >= 3; s >>= 1) {
    if (c < s) { red[c] += red[c + s]; red2[c] += red2[c + s]; }
    __syncthreads();
  }
  float mean = (red[0] + red[1] + red[2]) * (1.0f / 192.0f);
  float var  = (red2[0] + red2[1] + red2[2]) * (1.0f / 192.0f) - mean * mean;
  float rstd = rsqrtf(var + 1e-5f);
  out[(size_t)tok * 192 + c] = x1[(size_t)tok * 192 + c] + (v - mean) * rstd * g[c] + bt[c];
}

// ---------------- launcher ----------------
extern "C" void kernel_launch(void* const* d_in, const int* in_sizes, int n_in,
                              void* d_out, int out_size, void* d_ws, size_t ws_size,
                              hipStream_t stream) {
  (void)in_sizes; (void)n_in; (void)out_size; (void)ws_size;
  const float* x      = (const float*)d_in[0];
  const float* qkv_w  = (const float*)d_in[1];
  const float* qkv_b  = (const float*)d_in[2];
  const float* proj_w = (const float*)d_in[3];
  const float* proj_b = (const float*)d_in[4];
  const float* cpb_w1 = (const float*)d_in[5];
  const float* cpb_b1 = (const float*)d_in[6];
  const float* cpb_w2 = (const float*)d_in[7];
  const float* lsc    = (const float*)d_in[8];
  const float* n1g    = (const float*)d_in[9];
  const float* n1b    = (const float*)d_in[10];
  const float* n2g    = (const float*)d_in[11];
  const float* n2b    = (const float*)d_in[12];
  const float* fc1_w  = (const float*)d_in[13];
  const float* fc1_b  = (const float*)d_in[14];
  const float* fc2_w  = (const float*)d_in[15];
  const float* fc2_b  = (const float*)d_in[16];

  char* ws = (char*)d_ws;
  _Float16* qkvw16 = (_Float16*)(ws + O_QKVW);
  _Float16* projw16 = (_Float16*)(ws + O_PROJW);
  _Float16* fc1w16 = (_Float16*)(ws + O_FC1W);
  _Float16* fc2w16 = (_Float16*)(ws + O_FC2W);
  float*    tblf   = (float*)(ws + O_TBL);
  _Float16* q16    = (_Float16*)(ws + O_Q);
  _Float16* k16    = (_Float16*)(ws + O_K);
  _Float16* vT16   = (_Float16*)(ws + O_V);
  _Float16* xw16   = (_Float16*)(ws + O_B);
  _Float16* cat16  = (_Float16*)(ws + O_B);
  float*    projo  = (float*)(ws + O_E);
  float*    h2f    = (float*)(ws + O_E);
  float*    x1f    = (float*)(ws + O_F);
  _Float16* x1h    = (_Float16*)(ws + O_F16);
  _Float16* h1h    = (_Float16*)(ws + O_H1);
  float*    outp   = (float*)d_out;

  // 1. weights -> f16
  k_f32_to_f16<<<(576*192 + 255) / 256, 256, 0, stream>>>(qkv_w, qkvw16, 576*192);
  k_f32_to_f16<<<(192*192 + 255) / 256, 256, 0, stream>>>(proj_w, projw16, 192*192);
  k_f32_to_f16<<<(768*192 + 255) / 256, 256, 0, stream>>>(fc1_w, fc1w16, 768*192);
  k_f32_to_f16<<<(192*768 + 255) / 256, 256, 0, stream>>>(fc2_w, fc2w16, 192*768);
  // 2. continuous position-bias table
  k_cpb<<<(1575*6 + 127) / 128, 128, 0, stream>>>(cpb_w1, cpb_b1, cpb_w2, tblf);
  // 3. shift + window partition
  k_shift_partition<<<(NWIN*NTOK*C_) / 256, 256, 0, stream>>>(x, xw16);
  // 4. qkv GEMM (M=65536, N=576, K=192) with scatter epilogue
  k_gemm_wmma<3><<<dim3(512, 9), 256, 0, stream>>>(xw16, qkvw16, qkv_b,
      nullptr, nullptr, q16, k16, vT16, 576, 192);
  // 5. cosine-normalize q and k rows
  k_norm_qk<<<(2*NWIN*HEADS*NTOK + 255) / 256, 256, 0, stream>>>(q16, k16);
  // 6. fused attention (dyn LDS: q+k+v 48KB + P 128KB = 176KB)
  k_attn<<<NWIN * HEADS, 512, (size_t)(3*8192 + 65536) * 2, stream>>>(
      q16, k16, vT16, lsc, tblf, cat16);
  // 7. output projection (M=65536, N=192, K=192)
  k_gemm_wmma<0><<<dim3(512, 3), 256, 0, stream>>>(cat16, projw16, proj_b,
      projo, nullptr, nullptr, nullptr, nullptr, 192, 192);
  // 8. window reverse + roll + LN1 + residual
  k_ln1<<<MTOT, 192, 0, stream>>>(projo, x, n1g, n1b, x1f, x1h);
  // 9. fc1 + exact GELU (M=65536, N=768, K=192)
  k_gemm_wmma<2><<<dim3(512, 12), 256, 0, stream>>>(x1h, fc1w16, fc1_b,
      nullptr, h1h, nullptr, nullptr, nullptr, 768, 192);
  // 10. fc2 (M=65536, N=192, K=768)
  k_gemm_wmma<0><<<dim3(512, 3), 256, 0, stream>>>(h1h, fc2w16, fc2_b,
      h2f, nullptr, nullptr, nullptr, nullptr, 192, 768);
  // 11. LN2 + residual -> output
  k_ln2<<<MTOT, 192, 0, stream>>>(h2f, x1f, n2g, n2b, outp);
}
